// GraphSage_70093866271396
// MI455X (gfx1250) — compile-verified, matrix operand order checked by reading, not run on hardware
//
#include <hip/hip_runtime.h>

// ---------------------------------------------------------------------------
// Types for CDNA5 WMMA (wave32, 16x16x32 bf16 -> f32 accum)
// ---------------------------------------------------------------------------
typedef __bf16 bf16;
typedef __attribute__((ext_vector_type(8)))  bf16  v8bf;
typedef __attribute__((ext_vector_type(16))) bf16  v16bf;
typedef __attribute__((ext_vector_type(8)))  float v8f;
typedef __attribute__((__vector_size__(4 * sizeof(int)))) int i32x4;

#define F_IN 4096
#define HDIM 512
#define CDIM 6

// GEMM tiling: 256 threads = 8 waves; block tile 128x128, K step 32.
// Wave (waveM in 0..3, waveN in 0..1) computes 32 rows x 64 cols = 8 WMMAs.
#define BM 128
#define BN 128
#define TK 32

#define WMMA_BF16(a, b, c)                                                \
  __builtin_amdgcn_wmma_f32_16x16x32_bf16(false, (a), false, (b),         \
                                          (short)0, (c), false, false)

// ---- CDNA5 async global->LDS staging (ASYNCcnt path), with fallback -------
#if __has_builtin(__builtin_amdgcn_global_load_async_to_lds_b128)
#define USE_ASYNC_LDS 1
#else
#define USE_ASYNC_LDS 0
#endif

// Builtin signature (from hipcc diagnostic): arg0 = int4 addrspace(1)*,
// arg1 = int4 addrspace(3)*, then imm offset, imm cpol.
#define AS_GLOBAL_V4(p) ((__attribute__((address_space(1))) i32x4*)(p))
#define AS_LDS_V4(p)    ((__attribute__((address_space(3))) i32x4*)(p))

__device__ __forceinline__ void wait_async_le(int tag4) {
#if USE_ASYNC_LDS
#if __has_builtin(__builtin_amdgcn_s_wait_asynccnt)
  if (tag4) __builtin_amdgcn_s_wait_asynccnt(4);
  else      __builtin_amdgcn_s_wait_asynccnt(0);
#else
  if (tag4) asm volatile("s_wait_asynccnt 4" ::: "memory");
  else      asm volatile("s_wait_asynccnt 0" ::: "memory");
#endif
#else
  (void)tag4;
#endif
}

// ---------------------------------------------------------------------------
// fp32 -> bf16 convert / zero (grid-stride)
// ---------------------------------------------------------------------------
__global__ void f32_to_bf16_kernel(const float* __restrict__ in,
                                   bf16* __restrict__ out, size_t n) {
  size_t i = (size_t)blockIdx.x * blockDim.x + threadIdx.x;
  size_t stride = (size_t)gridDim.x * blockDim.x;
  for (; i < n; i += stride) out[i] = (bf16)in[i];
}

__global__ void zero_f32_kernel(float* __restrict__ p, size_t n) {
  size_t i = (size_t)blockIdx.x * blockDim.x + threadIdx.x;
  size_t stride = (size_t)gridDim.x * blockDim.x;
  for (; i < n; i += stride) p[i] = 0.0f;
}

// ---------------------------------------------------------------------------
// WMMA GEMM: Y[M x Nout] = A[M x K] * W^T, W row-major [Nout x K] (bf16).
// Double-buffered LDS, async-to-LDS staging, 8 WMMAs per wave per K-step.
// ---------------------------------------------------------------------------
__device__ __forceinline__ void stage_tile(const bf16* __restrict__ A,
                                           const bf16* __restrict__ W,
                                           bf16* As, bf16* Bs, int tid,
                                           int tileM, int tileN, int k0,
                                           int M, int Nout, int K) {
  // A tile: 128 rows x 32 bf16 = 512 x 16B chunks; same for B. 256 threads
  // move 2 chunks of each, coalesced over K (4 consecutive lanes per row).
#pragma unroll
  for (int i = 0; i < 2; ++i) {
    const int chunk = tid + i * 256;      // 0..511
    const int row   = chunk >> 2;         // 0..127
    const int cpos  = (chunk & 3) * 8;    // bf16 offset within row

    int ga = tileM + row; ga = (ga < M) ? ga : (M - 1);       // clamp tail
    int gb = tileN + row; gb = (gb < Nout) ? gb : (Nout - 1);
    const bf16* pa = A + (size_t)ga * K + k0 + cpos;
    const bf16* pw = W + (size_t)gb * K + k0 + cpos;
    bf16* la = As + row * TK + cpos;
    bf16* lb = Bs + row * TK + cpos;
#if USE_ASYNC_LDS
    __builtin_amdgcn_global_load_async_to_lds_b128(AS_GLOBAL_V4(pa), AS_LDS_V4(la), 0, 0);
    __builtin_amdgcn_global_load_async_to_lds_b128(AS_GLOBAL_V4(pw), AS_LDS_V4(lb), 0, 0);
#else
    const uint4 va = *(const uint4*)pa;   // grouped loads, then stores
    const uint4 vb = *(const uint4*)pw;
    *(uint4*)la = va;
    *(uint4*)lb = vb;
#endif
  }
}

__device__ __forceinline__ v16bf load_frag(const bf16* rowbase, int khalf) {
  // 16-bit fragment layout: lanes 0-15 hold K {0..7, 16..23},
  // lanes 16-31 hold K {8..15, 24..31} -> two 16B chunks per lane.
  const v8bf lo = *(const v8bf*)(rowbase + khalf * 8);
  const v8bf hi = *(const v8bf*)(rowbase + 16 + khalf * 8);
  return __builtin_shufflevector(lo, hi,
      0, 1, 2, 3, 4, 5, 6, 7, 8, 9, 10, 11, 12, 13, 14, 15);
}

__global__ __launch_bounds__(256)
void gemm_bf16_wmma_kernel(const bf16* __restrict__ A,
                           const bf16* __restrict__ W,
                           float* __restrict__ Y,
                           int M, int K, int Nout) {
  __shared__ __align__(16) bf16 As[2][BM * TK];
  __shared__ __align__(16) bf16 Bs[2][BN * TK];

  const int tid   = threadIdx.x;
  const int wave  = tid >> 5;
  const int lane  = tid & 31;
  const int l15   = lane & 15;
  const int khalf = lane >> 4;
  const int waveM = wave & 3;     // 4 M-slices of 32 rows
  const int waveN = wave >> 2;    // 2 N-slices of 64 cols
  const int tileM = blockIdx.y * BM;
  const int tileN = blockIdx.x * BN;

  v8f acc[2][4];
#pragma unroll
  for (int mi = 0; mi < 2; ++mi)
#pragma unroll
    for (int j = 0; j < 4; ++j) acc[mi][j] = (v8f){};

  const int nk = K / TK;
  stage_tile(A, W, As[0], Bs[0], tid, tileM, tileN, 0, M, Nout, K);

  for (int ki = 0; ki < nk; ++ki) {
    const int cur = ki & 1;
    const bool has_next = (ki + 1) < nk;
    if (has_next) {
      stage_tile(A, W, As[cur ^ 1], Bs[cur ^ 1], tid,
                 tileM, tileN, (ki + 1) * TK, M, Nout, K);
      wait_async_le(1);   // drain current slab's 4 DMAs, keep next 4 in flight
    } else {
      wait_async_le(0);
    }
    __syncthreads();

    const bf16* aBase = As[cur] + (waveM * 32) * TK;
    const v16bf a0 = load_frag(aBase + l15 * TK, khalf);
    const v16bf a1 = load_frag(aBase + (16 + l15) * TK, khalf);

#pragma unroll
    for (int j = 0; j < 4; ++j) {
      const bf16* bBase = Bs[cur] + (waveN * 64 + j * 16 + l15) * TK;
      const v16bf b = load_frag(bBase, khalf);
      acc[0][j] = WMMA_BF16(a0, b, acc[0][j]);
      acc[1][j] = WMMA_BF16(a1, b, acc[1][j]);
    }
    __syncthreads();   // compute done before this buffer is restaged
  }

  // C/D layout: VGPR r -> M = r + 8*(lane>=16), N = l15.
#pragma unroll
  for (int mi = 0; mi < 2; ++mi) {
#pragma unroll
    for (int j = 0; j < 4; ++j) {
      const int col   = tileN + waveN * 64 + j * 16 + l15;
      const int rbase = tileM + waveM * 32 + mi * 16 + khalf * 8;
#pragma unroll
      for (int r = 0; r < 8; ++r) {
        const int gm = rbase + r;
        if (gm < M) Y[(size_t)gm * Nout + col] = acc[mi][j][r];
      }
    }
  }
}

// ---------------------------------------------------------------------------
// Graph kernels: degree count, edge scatter-add (post-GEMM, low-dim), finalize
// ---------------------------------------------------------------------------
__global__ void degree_kernel(const long long* __restrict__ dst,
                              float* __restrict__ deg, int E) {
  int e = blockIdx.x * blockDim.x + threadIdx.x;
  if (e < E) atomicAdd(&deg[dst[e]], 1.0f);
}

__global__ void scatter_add_kernel(const float* __restrict__ feat,
                                   const long long* __restrict__ src,
                                   const long long* __restrict__ dst,
                                   float* __restrict__ acc, int F) {
  const int e = blockIdx.x;
  const long long s = src[e], d = dst[e];
  const float* fin  = feat + (size_t)s * F;
  float*       fout = acc  + (size_t)d * F;
  for (int f = threadIdx.x; f < F; f += blockDim.x)
    atomicAdd(&fout[f], fin[f]);
}

// out = [relu]( acc/max(deg,1) + bias + yr ); optional f32 and/or bf16 outputs
__global__ void finalize_kernel(const float* __restrict__ acc,
                                const float* __restrict__ deg,
                                const float* __restrict__ bias,
                                const float* __restrict__ yr,
                                float* __restrict__ out32,
                                bf16* __restrict__ out16,
                                int Nn, int F, int do_relu) {
  const size_t total = (size_t)Nn * F;
  size_t idx = (size_t)blockIdx.x * blockDim.x + threadIdx.x;
  const size_t stride = (size_t)gridDim.x * blockDim.x;
  for (; idx < total; idx += stride) {
    const int i = (int)(idx / F);
    const int f = (int)(idx % F);
    float d = deg[i]; d = (d > 1.0f) ? d : 1.0f;
    float v = acc[idx] / d + bias[f] + yr[idx];
    if (do_relu) v = fmaxf(v, 0.0f);
    if (out32) out32[idx] = v;
    if (out16) out16[idx] = (bf16)v;
  }
}

// Layer 3 (Hout=6): tiny GEMM pair done in plain fp32, one thread per (n,c).
__global__ void gemm_small_f32_kernel(const float* __restrict__ h,
                                      const float* __restrict__ Wl,
                                      const float* __restrict__ Wr,
                                      float* __restrict__ yl,
                                      float* __restrict__ yr,
                                      int Nn, int K, int C) {
  const int idx = blockIdx.x * blockDim.x + threadIdx.x;
  if (idx >= Nn * C) return;
  const int n = idx / C, c = idx % C;
  const float* hr = h  + (size_t)n * K;
  const float* wl = Wl + (size_t)c * K;
  const float* wr = Wr + (size_t)c * K;
  float sl = 0.0f, sr = 0.0f;
  for (int k = 0; k < K; ++k) {
    const float hv = hr[k];
    sl += hv * wl[k];
    sr += hv * wr[k];
  }
  yl[idx] = sl;
  yr[idx] = sr;
}

// ---------------------------------------------------------------------------
// Host-side orchestration
// ---------------------------------------------------------------------------
static inline char* bump(char*& p, size_t bytes) {
  char* r = p;
  p += (bytes + 255) & ~(size_t)255;
  return r;
}

extern "C" void kernel_launch(void* const* d_in, const int* in_sizes, int n_in,
                              void* d_out, int out_size, void* d_ws, size_t ws_size,
                              hipStream_t stream) {
  const float*      x    = (const float*)d_in[0];
  const long long*  ei   = (const long long*)d_in[1];
  const float*      W1l  = (const float*)d_in[2];
  const float*      b1   = (const float*)d_in[3];
  const float*      W1r  = (const float*)d_in[4];
  const float*      W2l  = (const float*)d_in[5];
  const float*      b2   = (const float*)d_in[6];
  const float*      W2r  = (const float*)d_in[7];
  const float*      W3l  = (const float*)d_in[8];
  const float*      b3   = (const float*)d_in[9];
  const float*      W3r  = (const float*)d_in[10];

  const int N = in_sizes[0] / F_IN;
  const int E = in_sizes[1] / 2;
  const long long* src = ei;
  const long long* dst = ei + E;

  // Workspace layout
  char* p = (char*)d_ws;
  bf16*  xb    = (bf16*)bump(p, (size_t)N * F_IN * sizeof(bf16));
  bf16*  w1lb  = (bf16*)bump(p, (size_t)HDIM * F_IN * sizeof(bf16));
  bf16*  w1rb  = (bf16*)bump(p, (size_t)HDIM * F_IN * sizeof(bf16));
  bf16*  w2lb  = (bf16*)bump(p, (size_t)HDIM * HDIM * sizeof(bf16));
  bf16*  w2rb  = (bf16*)bump(p, (size_t)HDIM * HDIM * sizeof(bf16));
  float* yl    = (float*)bump(p, (size_t)N * HDIM * sizeof(float));
  float* yr    = (float*)bump(p, (size_t)N * HDIM * sizeof(float));
  float* acc   = (float*)bump(p, (size_t)N * HDIM * sizeof(float));
  float* deg   = (float*)bump(p, (size_t)N * sizeof(float));
  bf16*  h1b   = (bf16*)bump(p, (size_t)N * HDIM * sizeof(bf16));
  float* yl3   = (float*)bump(p, (size_t)N * CDIM * sizeof(float));
  float* yr3   = (float*)bump(p, (size_t)N * CDIM * sizeof(float));
  (void)ws_size; (void)n_in; (void)out_size;

  float* h_out   = (float*)d_out;                       // [N, HDIM]
  float* cls_out = (float*)d_out + (size_t)N * HDIM;    // [N, CDIM]

  const dim3 blk256(256);
  const dim3 cvt_grid(4096);

  // --- 0) one-shot bf16 conversions -------------------------------------
  f32_to_bf16_kernel<<<cvt_grid, blk256, 0, stream>>>(x,   xb,   (size_t)N * F_IN);
  f32_to_bf16_kernel<<<cvt_grid, blk256, 0, stream>>>(W1l, w1lb, (size_t)HDIM * F_IN);
  f32_to_bf16_kernel<<<cvt_grid, blk256, 0, stream>>>(W1r, w1rb, (size_t)HDIM * F_IN);
  f32_to_bf16_kernel<<<cvt_grid, blk256, 0, stream>>>(W2l, w2lb, (size_t)HDIM * HDIM);
  f32_to_bf16_kernel<<<cvt_grid, blk256, 0, stream>>>(W2r, w2rb, (size_t)HDIM * HDIM);

  // --- degrees (shared by all layers) -----------------------------------
  zero_f32_kernel<<<256, blk256, 0, stream>>>(deg, (size_t)N);
  degree_kernel<<<(E + 255) / 256, blk256, 0, stream>>>(dst, deg, E);

  const dim3 gemm_blk(256);
  const dim3 gemm_grid(HDIM / BN, (N + BM - 1) / BM);

  // --- Layer 1: GEMM first, aggregate in 512-dim (8x less scatter traffic)
  gemm_bf16_wmma_kernel<<<gemm_grid, gemm_blk, 0, stream>>>(xb, w1lb, yl, N, F_IN, HDIM);
  gemm_bf16_wmma_kernel<<<gemm_grid, gemm_blk, 0, stream>>>(xb, w1rb, yr, N, F_IN, HDIM);
  zero_f32_kernel<<<2048, blk256, 0, stream>>>(acc, (size_t)N * HDIM);
  scatter_add_kernel<<<E, dim3(128), 0, stream>>>(yl, src, dst, acc, HDIM);
  finalize_kernel<<<2048, blk256, 0, stream>>>(acc, deg, b1, yr,
                                               (float*)nullptr, h1b, N, HDIM, 1);

  // --- Layer 2 ----------------------------------------------------------
  gemm_bf16_wmma_kernel<<<gemm_grid, gemm_blk, 0, stream>>>(h1b, w2lb, yl, N, HDIM, HDIM);
  gemm_bf16_wmma_kernel<<<gemm_grid, gemm_blk, 0, stream>>>(h1b, w2rb, yr, N, HDIM, HDIM);
  zero_f32_kernel<<<2048, blk256, 0, stream>>>(acc, (size_t)N * HDIM);
  scatter_add_kernel<<<E, dim3(128), 0, stream>>>(yl, src, dst, acc, HDIM);
  finalize_kernel<<<2048, blk256, 0, stream>>>(acc, deg, b2, yr,
                                               h_out, (bf16*)nullptr, N, HDIM, 1);

  // --- Layer 3 (tiny, fp32) ---------------------------------------------
  gemm_small_f32_kernel<<<(N * CDIM + 255) / 256, blk256, 0, stream>>>(
      h_out, W3l, W3r, yl3, yr3, N, HDIM, CDIM);
  zero_f32_kernel<<<256, blk256, 0, stream>>>(acc, (size_t)N * CDIM);
  scatter_add_kernel<<<E, dim3(32), 0, stream>>>(yl3, src, dst, acc, CDIM);
  finalize_kernel<<<256, blk256, 0, stream>>>(acc, deg, b3, yr3,
                                              cls_out, (bf16*)nullptr, N, CDIM, 0);
}
